// GroupedQueryAttention_74552042324378
// MI455X (gfx1250) — matrix operand order repaired and checked by reading, not compiled, not run on hardware
//
#include <hip/hip_runtime.h>
#include <hip/hip_bf16.h>

// ---------------------------------------------------------------------------
// GroupedQueryAttention forward for gfx1250 (MI455X), wave32 + WMMA f16.
// ---------------------------------------------------------------------------

typedef _Float16 half_t;
typedef __attribute__((ext_vector_type(16))) _Float16 v16h;
typedef __attribute__((ext_vector_type(8)))  _Float16 v8h;
typedef __attribute__((ext_vector_type(8)))  float    v8f;

static constexpr int D_MODEL   = 1024;
static constexpr int NUM_HEADS = 16;
static constexpr int NUM_GROUPS= 4;
static constexpr int HEAD_DIM  = 64;
static constexpr int KV_DIM    = 256;
static constexpr int BATCH     = 2;
static constexpr int SEQ       = 2048;

// ---------------------------------------------------------------------------
// WMMA helpers (wave32; A and B both use contiguous k = 16*(lane/16) + s)
// ---------------------------------------------------------------------------
__device__ __forceinline__ v8f wmma_f16(v16h a, v16h b, v8f c) {
  return __builtin_amdgcn_wmma_f32_16x16x32_f16(
      /*neg_a=*/false, a, /*neg_b=*/false, b,
      /*c_mod=*/(short)0, c, /*reuse_a=*/false, /*reuse_b=*/false);
}

// A-matrix 16x32 f16 fragment. base is row-major [.. x row_stride] halves.
// lanes 0-15: row m0+lane,   K = k0+0..15
// lanes16-31: row m0+lane-16,K = k0+16..31
__device__ __forceinline__ v16h load_a_frag(const half_t* base, int row_stride,
                                            int m0, int k0) {
  int lane = threadIdx.x & 31;
  int m  = m0 + (lane & 15);
  int kb = k0 + ((lane >> 4) << 4);
  const half_t* p = base + (size_t)m * row_stride + kb;
  v8h lo = *(const v8h*)(p);       // K = kb .. kb+7
  v8h hi = *(const v8h*)(p + 8);   // K = kb+8 .. kb+15
  v16h r;
#pragma unroll
  for (int i = 0; i < 8; ++i) { r[i] = lo[i]; r[8 + i] = hi[i]; }
  return r;
}

// B-matrix 32x16 f16 fragment where B[k][n] = base[n*row_stride + k]
// (i.e. base is the row-major [N x K] matrix whose transpose we multiply by).
// lanes 0-15: n = n0+lane,    K = k0+0..15  (contiguous 32B)
// lanes16-31: n = n0+lane-16, K = k0+16..31 (contiguous 32B)
__device__ __forceinline__ v16h load_b_frag(const half_t* base, int row_stride,
                                            int n0, int k0) {
  int lane = threadIdx.x & 31;
  int n  = n0 + (lane & 15);
  int kb = k0 + ((lane >> 4) << 4);
  return *(const v16h*)(base + (size_t)n * row_stride + kb);
}

// ---------------------------------------------------------------------------
// Elementwise f32 -> f16
// ---------------------------------------------------------------------------
__global__ void cvt_f32_to_f16(const float* __restrict__ in,
                               half_t* __restrict__ out, size_t n) {
  size_t i = (size_t)blockIdx.x * blockDim.x + threadIdx.x;
  if (i < n) out[i] = (half_t)in[i];
}

// ---------------------------------------------------------------------------
// Generic GEMM: C[M,N] = A[M,K] @ W[N,K]^T + bias  (f16 inputs, f32 accum)
// block = 128 threads (4 waves), wave w owns N-tile blockIdx.y*64 + w*16.
// Writes f16 (Ch) or f32 (Cf).
// ---------------------------------------------------------------------------
__global__ void gemm_nt_wmma(const half_t* __restrict__ A,
                             const half_t* __restrict__ W,
                             const float*  __restrict__ bias,
                             half_t* __restrict__ Ch,
                             float*  __restrict__ Cf,
                             int M, int N, int K) {
  int wave = threadIdx.x >> 5;
  int lane = threadIdx.x & 31;
  int m0 = blockIdx.x * 16;
  int n0 = blockIdx.y * 64 + wave * 16;
  if (n0 >= N) return;

  v8f acc = {};
  for (int k0 = 0; k0 < K; k0 += 32) {
    v16h af = load_a_frag(A, K, m0, k0);
    v16h bf = load_b_frag(W, K, n0, k0);
    acc = wmma_f16(af, bf, acc);
  }

  int n = n0 + (lane & 15);
  float bv = bias ? bias[n] : 0.0f;
#pragma unroll
  for (int r = 0; r < 8; ++r) {
    int m = m0 + r + ((lane >> 4) << 3);   // C layout: VGPR r -> M=r / r+8
    float val = acc[r] + bv;
    if (Ch) Ch[(size_t)m * N + n] = (half_t)val;
    else    Cf[(size_t)m * N + n] = val;
  }
}

// ---------------------------------------------------------------------------
// vT[b][g][d][s] = v[b][s][g*64+d]   (f16 -> f16, 1M elements)
// ---------------------------------------------------------------------------
__global__ void transpose_v(const half_t* __restrict__ hv,
                            half_t* __restrict__ hvT) {
  size_t i = (size_t)blockIdx.x * blockDim.x + threadIdx.x;
  const size_t total = (size_t)BATCH * NUM_GROUPS * HEAD_DIM * SEQ;
  if (i >= total) return;
  int s = (int)(i % SEQ);
  size_t t = i / SEQ;
  int d = (int)(t % HEAD_DIM); t /= HEAD_DIM;
  int g = (int)(t % NUM_GROUPS);
  int b = (int)(t / NUM_GROUPS);
  hvT[i] = hv[((size_t)(b * SEQ + s)) * KV_DIM + g * HEAD_DIM + d];
}

// ---------------------------------------------------------------------------
// Fused attention: one workgroup (4 waves / 128 threads) per (b,h,qtile16).
// Wave w owns keys [w*512, w*512+512).
// Dynamic LDS layout (bytes):
//   sc   f32[16][2048]       @ 0        (131072)  scores -> exp values
//   sp   f16[16][2048]       @ 131072   ( 65536)  normalized probs, f16
//   red  f32[16][8]          @ 196608   (   512)
//   rmax f32[16]             @ 197120
//   rsum f32[16]             @ 197184
//   pacc f32[4][16][64]      @ 197248   ( 16384)
// total 213632 bytes (within 320KB/WGP)
// ---------------------------------------------------------------------------
static constexpr size_t ATTN_SMEM = 213632;

__global__ void attention_wmma(const half_t* __restrict__ hq,   // [B,S,1024] f16
                               const half_t* __restrict__ hk,   // [B,S,256]  f16
                               const half_t* __restrict__ hvT,  // [B,G,64,S] f16
                               const float*  __restrict__ pos_bias, // [16,S,S]
                               const int*    __restrict__ mask,     // [B,S]
                               float*  __restrict__ attn_w,     // [B,16,S,S] f32 out
                               half_t* __restrict__ hao) {      // [B,S,1024] f16 out
  extern __shared__ char smem[];
  float*  sc   = (float*)(smem);
  half_t* sp   = (half_t*)(smem + 131072);
  float*  red  = (float*)(smem + 196608);
  float*  rmax = (float*)(smem + 197120);
  float*  rsum = (float*)(smem + 197184);
  float*  pacc = (float*)(smem + 197248);

  const int q0   = blockIdx.x * 16;
  const int h    = blockIdx.y;
  const int b    = blockIdx.z;
  const int g    = h / (NUM_HEADS / NUM_GROUPS);   // repeat_interleave mapping
  const int wave = threadIdx.x >> 5;
  const int lane = threadIdx.x & 31;

  const half_t* qbase = hq + ((size_t)b * SEQ) * D_MODEL + h * HEAD_DIM; // stride 1024
  const half_t* kbase = hk + ((size_t)b * SEQ) * KV_DIM  + g * HEAD_DIM; // stride 256

  // Q tile A-fragments (16 x 64 -> two 16x32 fragments), reused across keys
  v16h qa0 = load_a_frag(qbase, D_MODEL, q0, 0);
  v16h qa1 = load_a_frag(qbase, D_MODEL, q0, 32);

  const float scale = 0.125f;  // 1/sqrt(64)
  const int kstart = wave * 512;

  // ---- scores = QK^T * scale + pos_bias, masked; into LDS f32 ----
  for (int kc = 0; kc < 512; kc += 16) {
    const int key0 = kstart + kc;
    v16h kb0 = load_b_frag(kbase, KV_DIM, key0, 0);   // d = 0..31
    v16h kb1 = load_b_frag(kbase, KV_DIM, key0, 32);  // d = 32..63
    v8f c = {};
    c = wmma_f16(qa0, kb0, c);
    c = wmma_f16(qa1, kb1, c);
    const int n = key0 + (lane & 15);
#pragma unroll
    for (int r = 0; r < 8; ++r) {
      int qi = r + ((lane >> 4) << 3);                // 0..15 within tile
      float sv = c[r] * scale +
                 pos_bias[((size_t)h * SEQ + (q0 + qi)) * SEQ + n];
      if (mask[b * SEQ + q0 + qi] == 0) sv = -10000.0f;
      sc[qi * SEQ + n] = sv;
    }
  }
  __syncthreads();

  // ---- row max (stable even for fully-masked rows) ----
  {
    int row = threadIdx.x >> 3, seg = threadIdx.x & 7;
    float mx = -3.0e38f;
    for (int j = seg * 256; j < seg * 256 + 256; ++j)
      mx = fmaxf(mx, sc[row * SEQ + j]);
    red[row * 8 + seg] = mx;
  }
  __syncthreads();
  if (threadIdx.x < 16) {
    float mx = red[threadIdx.x * 8];
#pragma unroll
    for (int j = 1; j < 8; ++j) mx = fmaxf(mx, red[threadIdx.x * 8 + j]);
    rmax[threadIdx.x] = mx;
  }
  __syncthreads();

  // ---- exp + row sum ----
  {
    int row = threadIdx.x >> 3, seg = threadIdx.x & 7;
    float m = rmax[row];
    float s = 0.0f;
    for (int j = seg * 256; j < seg * 256 + 256; ++j) {
      float e = __expf(sc[row * SEQ + j] - m);
      sc[row * SEQ + j] = e;
      s += e;
    }
    red[row * 8 + seg] = s;
  }
  __syncthreads();
  if (threadIdx.x < 16) {
    float s = 0.0f;
#pragma unroll
    for (int j = 0; j < 8; ++j) s += red[threadIdx.x * 8 + j];
    rsum[threadIdx.x] = s;
  }
  __syncthreads();

  // ---- normalize: f32 attn_weights -> HBM, f16 probs -> LDS ----
  for (int t = threadIdx.x; t < 16 * SEQ; t += 128) {
    int row = t >> 11;           // /2048
    int col = t & (SEQ - 1);
    float w = sc[t] / rsum[row];
    sp[t] = (half_t)w;
    attn_w[(((size_t)(b * NUM_HEADS + h) * SEQ + (q0 + row)) * SEQ) + col] = w;
  }
  __syncthreads();

  // ---- attn @ V : wave accumulates its key range into 16x64 ----
  const half_t* vtb = hvT + ((size_t)(b * NUM_GROUPS + g) * HEAD_DIM) * SEQ;
  v8f oc0 = {}, oc1 = {}, oc2 = {}, oc3 = {};
  for (int kc = 0; kc < 512; kc += 32) {
    const int key0 = kstart + kc;
    v16h pa = load_a_frag(sp, SEQ, 0, key0);               // probs 16x32 (LDS)
    oc0 = wmma_f16(pa, load_b_frag(vtb, SEQ,  0, key0), oc0);
    oc1 = wmma_f16(pa, load_b_frag(vtb, SEQ, 16, key0), oc1);
    oc2 = wmma_f16(pa, load_b_frag(vtb, SEQ, 32, key0), oc2);
    oc3 = wmma_f16(pa, load_b_frag(vtb, SEQ, 48, key0), oc3);
  }
  {
    int n = lane & 15;
    int mo = (lane >> 4) << 3;
#pragma unroll
    for (int r = 0; r < 8; ++r) {
      int qi = r + mo;
      pacc[(wave * 16 + qi) * 64 + n]      = oc0[r];
      pacc[(wave * 16 + qi) * 64 + 16 + n] = oc1[r];
      pacc[(wave * 16 + qi) * 64 + 32 + n] = oc2[r];
      pacc[(wave * 16 + qi) * 64 + 48 + n] = oc3[r];
    }
  }
  __syncthreads();

  // ---- reduce 4 wave-partials, emit f16 attn_output ----
  for (int t = threadIdx.x; t < 16 * 64; t += 128) {
    float s = pacc[t] + pacc[1024 + t] + pacc[2048 + t] + pacc[3072 + t];
    int qi = t >> 6, d = t & 63;
    hao[((size_t)(b * SEQ) + q0 + qi) * D_MODEL + h * HEAD_DIM + d] = (half_t)s;
  }
}

// ---------------------------------------------------------------------------
// Host-side launch
// ---------------------------------------------------------------------------
extern "C" void kernel_launch(void* const* d_in, const int* in_sizes, int n_in,
                              void* d_out, int out_size, void* d_ws, size_t ws_size,
                              hipStream_t stream) {
  const float* query    = (const float*)d_in[0];
  const float* key      = (const float*)d_in[1];
  const float* value    = (const float*)d_in[2];
  const int*   mask     = (const int*)  d_in[3];
  const float* pos_bias = (const float*)d_in[4];
  const float* Wq = (const float*)d_in[5];
  const float* bq = (const float*)d_in[6];
  const float* Wk = (const float*)d_in[7];
  const float* bk = (const float*)d_in[8];
  const float* Wv = (const float*)d_in[9];
  const float* bv = (const float*)d_in[10];
  const float* Wo = (const float*)d_in[11];
  const float* bo = (const float*)d_in[12];

  float* out    = (float*)d_out;                         // [B,S,1024]
  float* attn_w = out + (size_t)BATCH * SEQ * D_MODEL;   // [B,16,S,S]

  // -------- workspace layout (f16 elements) --------
  const size_t nBSD = (size_t)BATCH * SEQ * D_MODEL;     // 4,194,304
  const size_t nBSK = (size_t)BATCH * SEQ * KV_DIM;      // 1,048,576
  const size_t nWq  = (size_t)D_MODEL * D_MODEL;         // 1,048,576
  const size_t nWk  = (size_t)KV_DIM * D_MODEL;          //   262,144

  half_t* p = (half_t*)d_ws;
  half_t* hWq = p; p += nWq;
  half_t* hWk = p; p += nWk;
  half_t* hWv = p; p += nWk;
  half_t* hWo = p; p += nWq;
  half_t* hXq = p; p += nBSD;   // query f16 (reused later as hao)
  half_t* hXk = p; p += nBSD;   // key f16   (reused later as hvT)
  half_t* hXv = p; p += nBSD;   // value f16
  half_t* hq  = p; p += nBSD;   // projected Q f16
  half_t* hk  = p; p += nBSK;   // projected K f16
  half_t* hv  = p; p += nBSK;   // projected V f16
  half_t* hao = hXq;            // attn output f16 (alias, hXq dead by then)
  half_t* hvT = hXk;            // transposed V    (alias, hXk dead by then)

  auto cvt = [&](const float* src, half_t* dst, size_t n) {
    cvt_f32_to_f16<<<dim3((unsigned)((n + 255) / 256)), 256, 0, stream>>>(src, dst, n);
  };
  cvt(Wq, hWq, nWq);
  cvt(Wk, hWk, nWk);
  cvt(Wv, hWv, nWk);
  cvt(Wo, hWo, nWq);
  cvt(query, hXq, nBSD);
  cvt(key,   hXk, nBSD);
  cvt(value, hXv, nBSD);

  const int M = BATCH * SEQ;  // 4096
  // Q projection: [4096,1024] = X @ Wq^T + bq
  gemm_nt_wmma<<<dim3(M / 16, D_MODEL / 64), 128, 0, stream>>>(
      hXq, hWq, bq, hq, nullptr, M, D_MODEL, D_MODEL);
  // K projection: [4096,256]
  gemm_nt_wmma<<<dim3(M / 16, KV_DIM / 64), 128, 0, stream>>>(
      hXk, hWk, bk, hk, nullptr, M, KV_DIM, D_MODEL);
  // V projection: [4096,256]
  gemm_nt_wmma<<<dim3(M / 16, KV_DIM / 64), 128, 0, stream>>>(
      hXv, hWv, bv, hv, nullptr, M, KV_DIM, D_MODEL);

  // V transpose per (b,g): [64][S]   (hXk is dead now)
  {
    const size_t n = (size_t)BATCH * NUM_GROUPS * HEAD_DIM * SEQ;
    transpose_v<<<dim3((unsigned)((n + 255) / 256)), 256, 0, stream>>>(hv, hvT);
  }

  // Fused attention (hXq is dead now -> hao)
  attention_wmma<<<dim3(SEQ / 16, NUM_HEADS, BATCH), 128, ATTN_SMEM, stream>>>(
      hq, hk, hvT, pos_bias, mask, attn_w, hao);

  // Output projection: out = hao @ Wo^T + bo (f32 result)
  gemm_nt_wmma<<<dim3(M / 16, D_MODEL / 64), 128, 0, stream>>>(
      hao, hWo, bo, nullptr, out, M, D_MODEL, D_MODEL);
}
